// SelfAttention_2456721293951
// MI455X (gfx1250) — matrix operand order repaired and checked by reading, not compiled, not run on hardware
//
#include <hip/hip_runtime.h>
#include <hip/hip_bf16.h>
#include <stdint.h>

// ---- CDNA5 wave32 WMMA / TDM types ----
typedef __attribute__((ext_vector_type(16))) __bf16 v16bf;
typedef __attribute__((ext_vector_type(8)))  __bf16 v8bf;
typedef __attribute__((ext_vector_type(8)))  float  v8f;
typedef __attribute__((ext_vector_type(4)))  unsigned int v4u;
typedef __attribute__((ext_vector_type(8)))  int v8i;
typedef __attribute__((ext_vector_type(4)))  int v4i;

#define BATCH 8
#define CCH   128
#define NPIX  4096   // 64*64
#define KPAD  32     // CQK=16 padded to WMMA K=32 with zeros

#if defined(__HIP_DEVICE_COMPILE__) && __has_builtin(__builtin_amdgcn_tensor_load_to_lds) && __has_builtin(__builtin_amdgcn_s_wait_tensorcnt)
#define USE_TDM 1
#else
#define USE_TDM 0
#endif

static __device__ __forceinline__ __bf16 f2bf(float f) { return (__bf16)f; }

// Combine two contiguous 8x bf16 chunks into a 16-bit A-fragment
// (A 16x32 layout: elems 0..15 -> K = 16*half+0..15, contiguous)
static __device__ __forceinline__ v16bf loadA2(const __bf16* p0, const __bf16* p1) {
  v8bf lo = *(const v8bf*)p0;
  v8bf hi = *(const v8bf*)p1;
  return __builtin_shufflevector(lo, hi, 0,1,2,3,4,5,6,7,8,9,10,11,12,13,14,15);
}

#if USE_TDM
// 2D tensor-DMA: global (rows x elements, bf16) -> LDS, per ISA 08_async_tensor D# layout.
static __device__ __forceinline__ void tdm_load_2d(unsigned lds_off, const void* gaddr,
                                                   unsigned tensor_d0, unsigned tensor_d1,
                                                   unsigned stride0, unsigned tile_d0,
                                                   unsigned tile_d1) {
  unsigned long long ga = (unsigned long long)(uintptr_t)gaddr;
  v4u g0;
  g0[0] = 1u;                                          // count=1, user mode
  g0[1] = lds_off;                                     // lds_addr (bytes)
  g0[2] = (unsigned)(ga & 0xFFFFFFFFu);                // global_addr[31:0]
  g0[3] = (unsigned)((ga >> 32) & 0x1FFFFFFu) | (2u << 30); // global_addr[56:32] | type=2
  v8i g1;
  g1[0] = (int)(1u << 16);                             // data_size=1 (2 bytes), mask=0
  g1[1] = (int)((tensor_d0 & 0xFFFFu) << 16);          // tensor_dim0[15:0]
  g1[2] = (int)(((tensor_d0 >> 16) & 0xFFFFu) | ((tensor_d1 & 0xFFFFu) << 16));
  g1[3] = (int)(((tensor_d1 >> 16) & 0xFFFFu) | ((tile_d0 & 0xFFFFu) << 16));
  g1[4] = (int)(tile_d1 & 0xFFFFu);                    // tile_dim1 ; tile_dim2=0
  g1[5] = (int)stride0;                                // tensor_dim0_stride[31:0]
  g1[6] = 0;
  g1[7] = 0;
  v4i z = {};
#if __clang_major__ >= 23
  v8i z8 = {};
  __builtin_amdgcn_tensor_load_to_lds(g0, g1, z, z, z8, 0);
#else
  __builtin_amdgcn_tensor_load_to_lds(g0, g1, z, z, 0);
#endif
}
template <typename T>
static __device__ __forceinline__ unsigned ldsOff(T* p) {
  return (unsigned)(uintptr_t)p;   // low 32 bits of flat LDS address = LDS byte offset
}
#endif

// =====================================================================
// Prep 1: transpose+convert x[B][C][N] f32 -> xT[B][N][C] bf16
// =====================================================================
__global__ void __launch_bounds__(256)
prep_x_kernel(const float* __restrict__ x, __bf16* __restrict__ xT) {
  const int idx = blockIdx.x * 256 + threadIdx.x;   // B*N*16 groups of 8 channels
  const int b   = idx >> 16;                        // 4096*16 = 65536 per batch
  const int rem = idx & 0xFFFF;
  const int n   = rem >> 4;
  const int g   = rem & 15;
  v8bf o;
#pragma unroll
  for (int e = 0; e < 8; ++e)
    o[e] = f2bf(x[((size_t)b * CCH + g * 8 + e) * NPIX + n]);
  *(v8bf*)&xT[((size_t)b * NPIX + n) * CCH + g * 8] = o;
}

// =====================================================================
// Prep 2: convert weights to bf16 (row-major, channel-contiguous)
// =====================================================================
__global__ void __launch_bounds__(256)
prep_w_kernel(const float* __restrict__ wq, const float* __restrict__ wk,
              const float* __restrict__ wv,
              __bf16* __restrict__ wqb, __bf16* __restrict__ wkb,
              __bf16* __restrict__ wvb) {
  const int t = blockIdx.x * 256 + threadIdx.x;
  if (t < 2048) { wqb[t] = f2bf(wq[t]); wkb[t] = f2bf(wk[t]); }
  const int u = t - 2048;
  if (u >= 0 && u < 16384) wvb[u] = f2bf(wv[u]);
}

// =====================================================================
// Kernel A: 1x1-conv projections via bf16 WMMA, all-vector fragment loads.
// One wave per (batch, 16-wide n-tile): 40 WMMAs.
// =====================================================================
__global__ void __launch_bounds__(256)
qkv_proj_kernel(const __bf16* __restrict__ xT,
                const __bf16* __restrict__ wqb, const float* __restrict__ bq,
                const __bf16* __restrict__ wkb, const float* __restrict__ bk,
                const __bf16* __restrict__ wvb, const float* __restrict__ bv,
                __bf16* __restrict__ qpad, __bf16* __restrict__ kpad,
                __bf16* __restrict__ vbf) {
  const int lane  = threadIdx.x & 31;
  const int wid   = threadIdx.x >> 5;
  const int task  = blockIdx.x * 8 + wid;          // B * N/16 = 2048 tasks
  const int b     = task >> 8;
  const int n0    = (task & 255) << 4;
  const int lhalf = (lane >= 16) ? 8 : 0;
  const int lmod  = lane & 15;
  const int kb    = (lane >= 16) ? 16 : 0;
  const int n     = n0 + lmod;

  v8f accQ = {}; v8f accK = {}; v8f accV[8];
#pragma unroll
  for (int t = 0; t < 8; ++t) accV[t] = (v8f){};

  const __bf16* xrow = xT + ((size_t)b * NPIX + n) * CCH;
#pragma unroll
  for (int ks = 0; ks < 4; ++ks) {
    const int kbase = ks * 32;
    const v16bf bx = *(const v16bf*)(xrow + kbase + kb);                  // B-frag
    const v16bf aq = loadA2(wqb + lmod * CCH + kbase + kb,
                            wqb + lmod * CCH + kbase + kb + 8);
    const v16bf ak = loadA2(wkb + lmod * CCH + kbase + kb,
                            wkb + lmod * CCH + kbase + kb + 8);
    accQ = __builtin_amdgcn_wmma_f32_16x16x32_bf16(false, aq, false, bx, (short)0, accQ, false, false);
    accK = __builtin_amdgcn_wmma_f32_16x16x32_bf16(false, ak, false, bx, (short)0, accK, false, false);
#pragma unroll
    for (int ct = 0; ct < 8; ++ct) {
      const __bf16* wrow = wvb + (size_t)(ct * 16 + lmod) * CCH + kbase;
      const v16bf av = loadA2(wrow + kb, wrow + kb + 8);
      accV[ct] = __builtin_amdgcn_wmma_f32_16x16x32_bf16(false, av, false, bx, (short)0, accV[ct], false, false);
    }
  }

  const __bf16 bz = f2bf(0.0f);
#pragma unroll
  for (int r = 0; r < 8; ++r) {
    const int c  = r + lhalf;
    const size_t qi = ((size_t)b * NPIX + n) * KPAD + c;
    qpad[qi]      = f2bf(accQ[r] + bq[c]);
    qpad[qi + 16] = bz;
    kpad[qi]      = f2bf(accK[r] + bk[c]);
    kpad[qi + 16] = bz;
#pragma unroll
    for (int ct = 0; ct < 8; ++ct) {
      const int cc = ct * 16 + c;
      vbf[((size_t)b * CCH + cc) * NPIX + n] = f2bf(accV[ct][r] + bv[cc]);
    }
  }
}

// =====================================================================
// Kernel B: fused flash attention. 8 waves/WG, one 16-row i-tile per wave.
// K and V panels staged in LDS (TDM tensor_load_to_lds, double-buffered),
// shared by all 8 waves. Two-pass softmax, all GEMMs on bf16 WMMA.
// =====================================================================
__global__ void __launch_bounds__(256)
flash_attn_kernel(const float* __restrict__ x,
                  const __bf16* __restrict__ qpad,
                  const __bf16* __restrict__ kpad,
                  const __bf16* __restrict__ vbf,
                  float* __restrict__ out) {
  __shared__ __attribute__((aligned(32))) __bf16 kTile[2][128][KPAD];  // pass-1 K panels
  __shared__ __attribute__((aligned(32))) __bf16 k2[2][32][KPAD];      // pass-2 K chunks
  __shared__ __attribute__((aligned(32))) __bf16 v2[2][CCH][32];       // pass-2 V panels
  __shared__ __attribute__((aligned(32))) __bf16 pTile[8][16][32];     // per-wave P^T
  __shared__ float rowSumLds[8][16];

  const int lane  = threadIdx.x & 31;
  const int w     = threadIdx.x >> 5;
  const int b     = blockIdx.x >> 5;
  const int itile = blockIdx.x & 31;
  const int i0    = itile * 128 + w * 16;
  const int lhalf = (lane >= 16) ? 8 : 0;
  const int lmod  = lane & 15;
  const int kb    = (lane >= 16) ? 16 : 0;

  const __bf16* kbp   = kpad + (size_t)b * NPIX * KPAD;
  const __bf16* vbase = vbf + (size_t)b * CCH * NPIX;

  // Q A-fragment for this wave's 16 rows
  const __bf16* qrow = qpad + ((size_t)b * NPIX + i0 + lmod) * KPAD;
  const v16bf aq = loadA2(qrow + kb, qrow + kb + 8);

  // ---- Pass 1: row max over all j ----
  float rm[8];
#pragma unroll
  for (int r = 0; r < 8; ++r) rm[r] = -3.0e38f;

#if USE_TDM
  if (w == 0)
    tdm_load_2d(ldsOff(&kTile[0][0][0]), kbp, KPAD, NPIX, KPAD, KPAD, 128);
#endif
  int buf = 0;
  for (int Jb = 0; Jb < NPIX; Jb += 128, buf ^= 1) {
#if USE_TDM
    if (w == 0) __builtin_amdgcn_s_wait_tensorcnt(0);
    __syncthreads();
    if (w == 0 && Jb + 128 < NPIX)
      tdm_load_2d(ldsOff(&kTile[buf ^ 1][0][0]), kbp + (size_t)(Jb + 128) * KPAD,
                  KPAD, NPIX, KPAD, KPAD, 128);
#else
    __syncthreads();
    {
      const int t = threadIdx.x, row = t >> 1, h16 = (t & 1) * 16;
      *(v16bf*)&kTile[buf][row][h16] =
          *(const v16bf*)(kbp + (size_t)(Jb + row) * KPAD + h16);
    }
    __syncthreads();
#endif
#pragma unroll
    for (int js = 0; js < 8; ++js) {
      const v16bf bkf = *(const v16bf*)&kTile[buf][js * 16 + lmod][kb];
      v8f z = {};
      v8f s = __builtin_amdgcn_wmma_f32_16x16x32_bf16(false, aq, false, bkf, (short)0, z, false, false);
#pragma unroll
      for (int r = 0; r < 8; ++r) rm[r] = fmaxf(rm[r], s[r]);
    }
  }
#pragma unroll
  for (int m = 1; m < 16; m <<= 1)
#pragma unroll
    for (int r = 0; r < 8; ++r) rm[r] = fmaxf(rm[r], __shfl_xor(rm[r], m, 32));

  // ---- Pass 2: exp + output accumulation ----
  float rs[8];
#pragma unroll
  for (int r = 0; r < 8; ++r) rs[r] = 0.0f;
  v8f accO[8];
#pragma unroll
  for (int t = 0; t < 8; ++t) accO[t] = (v8f){};

#if USE_TDM
  if (w == 0) {
    tdm_load_2d(ldsOff(&k2[0][0][0]), kbp, KPAD, NPIX, KPAD, KPAD, 32);
    tdm_load_2d(ldsOff(&v2[0][0][0]), vbase, NPIX, CCH, NPIX, 32, CCH);
  }
#endif
  buf = 0;
  for (int jj = 0; jj < NPIX; jj += 32, buf ^= 1) {
#if USE_TDM
    if (w == 0) __builtin_amdgcn_s_wait_tensorcnt(0);
    __syncthreads();
    if (w == 0 && jj + 32 < NPIX) {
      tdm_load_2d(ldsOff(&k2[buf ^ 1][0][0]), kbp + (size_t)(jj + 32) * KPAD,
                  KPAD, NPIX, KPAD, KPAD, 32);
      tdm_load_2d(ldsOff(&v2[buf ^ 1][0][0]), vbase + (jj + 32),
                  NPIX, CCH, NPIX, 32, CCH);
    }
#else
    __syncthreads();
    {
      const int t = threadIdx.x, row = t >> 1, h16 = (t & 1) * 16;
      if (t < 64)
        *(v16bf*)&k2[buf][row][h16] =
            *(const v16bf*)(kbp + (size_t)(jj + row) * KPAD + h16);
      *(v16bf*)&v2[buf][row][h16] =
          *(const v16bf*)(vbase + (size_t)row * NPIX + jj + h16);
    }
    __syncthreads();
#endif
#pragma unroll
    for (int t = 0; t < 2; ++t) {
      const v16bf bkf = *(const v16bf*)&k2[buf][t * 16 + lmod][kb];
      v8f z = {};
      v8f s = __builtin_amdgcn_wmma_f32_16x16x32_bf16(false, aq, false, bkf, (short)0, z, false, false);
#pragma unroll
      for (int r = 0; r < 8; ++r) {
        const float p = __expf(s[r] - rm[r]);            // unnormalized softmax
        rs[r] += p;
        pTile[w][r + lhalf][t * 16 + lmod] = f2bf(p);    // transpose via LDS
      }
    }
    const v16bf pb = *(const v16bf*)&pTile[w][lmod][kb]; // P^T B-fragment
#pragma unroll
    for (int ct = 0; ct < 8; ++ct) {
      const __bf16* vrow = &v2[buf][ct * 16 + lmod][0];
      const v16bf av = loadA2(vrow + kb, vrow + kb + 8);
      accO[ct] = __builtin_amdgcn_wmma_f32_16x16x32_bf16(false, av, false, pb, (short)0, accO[ct], false, false);
    }
  }

  // row sums -> per-i broadcast through LDS
#pragma unroll
  for (int m = 1; m < 16; m <<= 1)
#pragma unroll
    for (int r = 0; r < 8; ++r) rs[r] += __shfl_xor(rs[r], m, 32);
  if (lane == 0) {
#pragma unroll
    for (int r = 0; r < 8; ++r) rowSumLds[w][r] = rs[r];
  }
  if (lane == 16) {
#pragma unroll
    for (int r = 0; r < 8; ++r) rowSumLds[w][8 + r] = rs[r];
  }
  const float rinv = 1.0f / rowSumLds[w][lmod];

  // Epilogue: normalize, residual add, coalesced along i
  const int i = i0 + lmod;
#pragma unroll
  for (int ct = 0; ct < 8; ++ct) {
#pragma unroll
    for (int r = 0; r < 8; ++r) {
      const int c = ct * 16 + r + lhalf;
      const size_t idx = ((size_t)b * CCH + c) * NPIX + i;
      out[idx] = accO[ct][r] * rinv + x[idx];
    }
  }
}

extern "C" void kernel_launch(void* const* d_in, const int* in_sizes, int n_in,
                              void* d_out, int out_size, void* d_ws, size_t ws_size,
                              hipStream_t stream) {
  const float* x  = (const float*)d_in[0];
  const float* wq = (const float*)d_in[1];
  const float* bq = (const float*)d_in[2];
  const float* wk = (const float*)d_in[3];
  const float* bk = (const float*)d_in[4];
  const float* wv = (const float*)d_in[5];
  const float* bv = (const float*)d_in[6];
  float* out = (float*)d_out;

  __bf16* qpad = (__bf16*)d_ws;                               // 2 MB
  __bf16* kpad = qpad + (size_t)BATCH * NPIX * KPAD;          // 2 MB
  __bf16* vbf  = kpad + (size_t)BATCH * NPIX * KPAD;          // 8 MB
  __bf16* xT   = vbf  + (size_t)BATCH * CCH * NPIX;           // 8 MB
  __bf16* wqb  = xT   + (size_t)BATCH * NPIX * CCH;           // 4 KB
  __bf16* wkb  = wqb + 16 * CCH;
  __bf16* wvb  = wkb + 16 * CCH;                              // 32 KB

  prep_x_kernel<<<2048, 256, 0, stream>>>(x, xT);
  prep_w_kernel<<<72, 256, 0, stream>>>(wq, wk, wv, wqb, wkb, wvb);
  qkv_proj_kernel<<<256, 256, 0, stream>>>(xT, wqb, bq, wkb, bk, wvb, bv, qpad, kpad, vbf);
  flash_attn_kernel<<<256, 256, 0, stream>>>(x, qpad, kpad, vbf, out);
}
